// SlayerMBConv_80994493268063
// MI455X (gfx1250) — matrix operand order repaired and checked by reading, not compiled, not run on hardware
//
#include <hip/hip_runtime.h>

// ---------------------------------------------------------------------------
// MI455X (gfx1250) implementation of the sigma-delta MBConv block.
// 1x1 convs -> bf16 WMMA GEMMs (v_wmma_f32_16x16x32_bf16), everything else
// fused into lightweight bandwidth-bound kernels. Intermediates live in a
// [P][C] channel-last layout sized to stay resident in the 192MB L2.
// ---------------------------------------------------------------------------

typedef __bf16 bf16;
typedef __attribute__((ext_vector_type(16))) __bf16 v16bf;
typedef __attribute__((ext_vector_type(8)))  __bf16 v8bf;
typedef __attribute__((ext_vector_type(8)))  float  v8f;

#define NB   4          // batch
#define HW_  32         // H == W
#define TT   32         // timesteps
#define CIN  32
#define HID  128
#define SQZ  32
#define COUT 64
#define NHW  (NB*HW_*HW_)       // 4096 spatial positions
#define PTOT (NHW*TT)           // 131072 pixels (n,h,w,t) -> GEMM M dimension

__device__ __forceinline__ float pwl_silu(float x) {
    return x > 0.f ? x
         : (x > -1.f ? 0.25f * x
         : (x > -2.f ? -0.25f * (x + 2.f) : 0.f));
}

// ---------------------------------------------------------------------------
// Weight prep: weight-norm -> 8-bit quantize (scale 64) -> bf16, stored
// pre-swizzled in WMMA B-fragment order so GEMM lanes load one aligned v16bf.
// B layout used here (contiguous k order, same as A): lane = (co&15) | (16 if (k&31)>=16), element i
// within the v16bf is k&15.
// ---------------------------------------------------------------------------
template<int CI, int CO>
__global__ void prep_w1x1(const float* __restrict__ v, const float* __restrict__ g,
                          bf16* __restrict__ WqB) {
    __shared__ float red[CI];
    const int co = blockIdx.x;
    const int ci = threadIdx.x;
    float val = v[co * CI + ci];
    red[ci] = val * val;
    __syncthreads();
    for (int s = CI / 2; s > 0; s >>= 1) {
        if (ci < s) red[ci] += red[ci + s];
        __syncthreads();
    }
    float w = g[co] * val / sqrtf(red[0]);
    float q = fminf(fmaxf(rintf(w * 64.f), -128.f), 127.f) * 0.015625f;
    const int kb   = ci >> 5;
    const int ct   = co >> 4;
    const int lane = (co & 15) | (ci & 16);
    const int i    = ci & 15;
    WqB[(((kb * (CO / 16) + ct) * 32 + lane) << 4) + i] = (bf16)q;
}

__global__ void prep_wdw(const float* __restrict__ v, const float* __restrict__ g,
                         float* __restrict__ wdw) {
    const int c = threadIdx.x;
    if (c >= HID) return;
    float t[9], ss = 0.f;
    #pragma unroll
    for (int k = 0; k < 9; ++k) { t[k] = v[c * 9 + k]; ss += t[k] * t[k]; }
    const float inv_n = 1.f / sqrtf(ss);
    #pragma unroll
    for (int k = 0; k < 9; ++k) {
        float w = g[c] * t[k] * inv_n;
        wdw[c * 9 + k] = fminf(fmaxf(rintf(w * 64.f), -128.f), 127.f) * 0.015625f;
    }
}

// x [N][C][H][W][T] f32  ->  Xbf [(n,h,w,t)][C] bf16 (GEMM A operand layout)
__global__ void prep_input(const float* __restrict__ x, bf16* __restrict__ Xb) {
    const int idx = blockIdx.x * blockDim.x + threadIdx.x;
    if (idx >= NB * CIN * HW_ * HW_ * TT) return;
    const int t = idx & 31, w = (idx >> 5) & 31, h = (idx >> 10) & 31;
    const int c = (idx >> 15) & 31, n = idx >> 20;
    const long p = ((long)(n * 32 + h) * 32 + w) * 32 + t;
    Xb[p * CIN + c] = (bf16)x[idx];
}

// ---------------------------------------------------------------------------
// GEMM: Y[p][co] = sum_k Xbf[p][k] * W[k][co] via v_wmma_f32_16x16x32_bf16.
// One wave per 16-pixel M-tile; A fragment reused across all Co-tiles.
// Exact grid => EXEC all-ones (WMMA requirement).
// ---------------------------------------------------------------------------
template<int K, int CO>
__global__ void __launch_bounds__(256) gemm_wmma(const bf16* __restrict__ X,
                                                 const bf16* __restrict__ WqB,
                                                 float* __restrict__ Y) {
    constexpr int KB  = K / 32;
    constexpr int CoT = CO / 16;
    const int lane = threadIdx.x & 31;
    const int wave = threadIdx.x >> 5;
    const int l15  = lane & 15;
    const int half = lane >> 4;
    const long mTile = (long)blockIdx.x * 8 + wave;
    const bf16* __restrict__ Arow = X + (mTile * 16 + l15) * K;
    const v16bf* __restrict__ Bp  = (const v16bf*)WqB;

    v8f acc[CoT] = {};
    #pragma unroll
    for (int kb = 0; kb < KB; ++kb) {
        const int aoff = kb * 32 + half * 16;
        if (kb + 1 < KB) __builtin_prefetch(Arow + aoff + 32, 0, 3);
        v8bf a0 = *(const v8bf*)(Arow + aoff);        // K = off..off+7
        v8bf a1 = *(const v8bf*)(Arow + aoff + 8);    // K = off+8..off+15
        v16bf a = __builtin_shufflevector(a0, a1,
                    0, 1, 2, 3, 4, 5, 6, 7, 8, 9, 10, 11, 12, 13, 14, 15);
        #pragma unroll
        for (int ct = 0; ct < CoT; ++ct) {
            v16bf b = Bp[(kb * CoT + ct) * 32 + lane];
            acc[ct] = __builtin_amdgcn_wmma_f32_16x16x32_bf16(
                false, a, false, b, (short)0, acc[ct], false, false);
        }
    }
    // D layout: VGPR r -> (M = r + 8*half, N = lane&15) for each Co-tile
    float* __restrict__ yp0 = Y + (mTile * 16 + half * 8) * CO + l15;
    #pragma unroll
    for (int ct = 0; ct < CoT; ++ct) {
        #pragma unroll
        for (int r = 0; r < 8; ++r)
            yp0[(long)r * CO + ct * 16] = acc[ct][r];
    }
}

// ---------------------------------------------------------------------------
// Depthwise 3x3 (pad 1) over [P][HID] channel-last layout; t is innermost in p.
// ---------------------------------------------------------------------------
__global__ void dwconv3x3(const float* __restrict__ in, const float* __restrict__ wdw,
                          float* __restrict__ out) {
    const long idx = (long)blockIdx.x * blockDim.x + threadIdx.x; // over P*HID
    const int  c = (int)(idx & 127);
    const long p = idx >> 7;
    const int t = (int)(p & 31), w = (int)((p >> 5) & 31);
    const int h = (int)((p >> 10) & 31), n = (int)(p >> 15);
    const float* __restrict__ wc = wdw + c * 9;
    float s = 0.f;
    #pragma unroll
    for (int i = 0; i < 3; ++i) {
        const int hh = h + i - 1;
        if ((unsigned)hh >= 32u) continue;
        #pragma unroll
        for (int j = 0; j < 3; ++j) {
            const int ww = w + j - 1;
            if ((unsigned)ww >= 32u) continue;
            const long q = (((long)(n * 32 + hh) * 32 + ww) * 32 + t);
            s += wc[i * 3 + j] * in[(q << 7) + c];
        }
    }
    out[(p << 7) + c] = s;
}

// ---------------------------------------------------------------------------
// Per-channel column sums for mean-only BN (LDS partials + global atomics).
// ---------------------------------------------------------------------------
__global__ void zero_f32(float* __restrict__ p, int n) {
    int i = blockIdx.x * blockDim.x + threadIdx.x;
    if (i < n) p[i] = 0.f;
}

template<int C>
__global__ void colsum(const float* __restrict__ Y, float* __restrict__ sums) {
    __shared__ float sm[C];
    if (threadIdx.x < C) sm[threadIdx.x] = 0.f;
    __syncthreads();
    constexpr int RPB = 256 / C;              // rows handled per block per step
    const int c = threadIdx.x & (C - 1);
    const int r = threadIdx.x / C;
    float s = 0.f;
    for (int p = blockIdx.x * RPB + r; p < PTOT; p += gridDim.x * RPB)
        s += Y[(long)p * C + c];
    atomicAdd(&sm[c], s);
    __syncthreads();
    if (threadIdx.x < C) atomicAdd(&sums[threadIdx.x], sm[threadIdx.x]);
}

// ---------------------------------------------------------------------------
// Fused epilogue: mean subtract -> cumsum over T -> pwl_silu -> delta
// (+ optional residual add), emitting f32 / bf16 / final-NCHWT as requested.
// One thread owns one (n,h,w,c) and walks T sequentially (t innermost in p).
// ---------------------------------------------------------------------------
template<int C>
__global__ void epilogue(const float* __restrict__ Y, const float* __restrict__ sums,
                         const float* __restrict__ res, float* __restrict__ outF,
                         bf16* __restrict__ outB, float* __restrict__ outFinal) {
    const int idx = blockIdx.x * blockDim.x + threadIdx.x;
    if (idx >= NHW * C) return;
    const int c   = idx & (C - 1);
    const int pos = idx / C;                  // (n*32 + h)*32 + w
    const long pb = (long)pos * TT;
    const float m = sums[c] * (1.f / (float)PTOT);
    float s = 0.f, aprev = 0.f;
    for (int t = 0; t < TT; ++t) {
        const long o = (pb + t) * C + c;
        s += Y[o] - m;
        const float a = pwl_silu(s);
        float d = a - aprev;                  // t==0: aprev==0 -> d = a[0]
        aprev = a;
        if (res)  d += res[o];
        if (outF) outF[o] = d;
        if (outB) outB[o] = (bf16)d;
        if (outFinal) {
            const int n = pos >> 10, hw = pos & 1023;
            outFinal[(((long)(n * C + c)) * 1024 + hw) * TT + t] = d;
        }
    }
}

// ---------------------------------------------------------------------------
extern "C" void kernel_launch(void* const* d_in, const int* in_sizes, int n_in,
                              void* d_out, int out_size, void* d_ws, size_t ws_size,
                              hipStream_t stream) {
    const float* x     = (const float*)d_in[0];
    const float* v_in  = (const float*)d_in[1];
    const float* g_in  = (const float*)d_in[2];
    const float* v_dw  = (const float*)d_in[3];
    const float* g_dw  = (const float*)d_in[4];
    const float* v_sq  = (const float*)d_in[5];
    const float* g_sq  = (const float*)d_in[6];
    const float* v_ex  = (const float*)d_in[7];
    const float* g_ex  = (const float*)d_in[8];
    const float* v_out = (const float*)d_in[9];
    const float* g_out = (const float*)d_in[10];
    float* out = (float*)d_out;

    char* ws = (char*)d_ws;
    size_t off = 0;
    auto alloc = [&](size_t bytes) -> void* {
        off = (off + 255) & ~(size_t)255;
        void* p = ws + off;
        off += bytes;
        return p;
    };
    bf16*  Xbf  = (bf16*) alloc((size_t)PTOT * HID * sizeof(bf16));  // GEMM input
    float* Yf   = (float*)alloc((size_t)PTOT * HID * sizeof(float)); // conv out
    float* resF = (float*)alloc((size_t)PTOT * HID * sizeof(float)); // block1 out / residual
    float* sums = (float*)alloc(HID * sizeof(float));
    bf16*  Win  = (bf16*) alloc((size_t)CIN * HID * sizeof(bf16));
    bf16*  Wsq  = (bf16*) alloc((size_t)HID * SQZ * sizeof(bf16));
    bf16*  Wex  = (bf16*) alloc((size_t)SQZ * HID * sizeof(bf16));
    bf16*  Wpr  = (bf16*) alloc((size_t)HID * COUT * sizeof(bf16));
    float* Wdw  = (float*)alloc((size_t)HID * 9 * sizeof(float));

    // --- weight prep (tiny) + input layout transform --------------------
    prep_w1x1<CIN, HID><<<HID,  CIN, 0, stream>>>(v_in,  g_in,  Win);
    prep_w1x1<HID, SQZ><<<SQZ,  HID, 0, stream>>>(v_sq,  g_sq,  Wsq);
    prep_w1x1<SQZ, HID><<<HID,  SQZ, 0, stream>>>(v_ex,  g_ex,  Wex);
    prep_w1x1<HID, COUT><<<COUT, HID, 0, stream>>>(v_out, g_out, Wpr);
    prep_wdw<<<1, 128, 0, stream>>>(v_dw, g_dw, Wdw);
    prep_input<<<(NB * CIN * HW_ * HW_ * TT) / 256, 256, 0, stream>>>(x, Xbf);

    const int gemmBlocks = PTOT / 128;   // 8 waves/block, 16-pixel M-tile/wave

    // --- block 1: 1x1 expand (32 -> 128), output kept f32 as residual ---
    gemm_wmma<CIN, HID><<<gemmBlocks, 256, 0, stream>>>(Xbf, Win, Yf);
    zero_f32<<<1, HID, 0, stream>>>(sums, HID);
    colsum<HID><<<512, 256, 0, stream>>>(Yf, sums);
    epilogue<HID><<<(NHW * HID) / 256, 256, 0, stream>>>(
        Yf, sums, nullptr, resF, nullptr, nullptr);

    // --- block 2: 3x3 depthwise on 128 channels -------------------------
    dwconv3x3<<<((long)PTOT * HID) / 256, 256, 0, stream>>>(resF, Wdw, Yf);
    zero_f32<<<1, HID, 0, stream>>>(sums, HID);
    colsum<HID><<<512, 256, 0, stream>>>(Yf, sums);
    epilogue<HID><<<(NHW * HID) / 256, 256, 0, stream>>>(
        Yf, sums, nullptr, nullptr, Xbf, nullptr);

    // --- block 3: 1x1 squeeze (128 -> 32) -------------------------------
    gemm_wmma<HID, SQZ><<<gemmBlocks, 256, 0, stream>>>(Xbf, Wsq, Yf);
    zero_f32<<<1, SQZ, 0, stream>>>(sums, SQZ);
    colsum<SQZ><<<512, 256, 0, stream>>>(Yf, sums);
    epilogue<SQZ><<<(NHW * SQZ) / 256, 256, 0, stream>>>(
        Yf, sums, nullptr, nullptr, Xbf, nullptr);

    // --- block 4: 1x1 excite (32 -> 128) + residual add ------------------
    gemm_wmma<SQZ, HID><<<gemmBlocks, 256, 0, stream>>>(Xbf, Wex, Yf);
    zero_f32<<<1, HID, 0, stream>>>(sums, HID);
    colsum<HID><<<512, 256, 0, stream>>>(Yf, sums);
    epilogue<HID><<<(NHW * HID) / 256, 256, 0, stream>>>(
        Yf, sums, resF, nullptr, Xbf, nullptr);

    // --- block 5: 1x1 project (128 -> 64), final NCHWT store -------------
    gemm_wmma<HID, COUT><<<gemmBlocks, 256, 0, stream>>>(Xbf, Wpr, Yf);
    zero_f32<<<1, COUT, 0, stream>>>(sums, COUT);
    colsum<COUT><<<512, 256, 0, stream>>>(Yf, sums);
    epilogue<COUT><<<(NHW * COUT) / 256, 256, 0, stream>>>(
        Yf, sums, nullptr, nullptr, nullptr, out);
}
